// PointerGenerator_57062935495095
// MI455X (gfx1250) — matrix operand order repaired and checked by reading, not compiled, not run on hardware
//
#include <hip/hip_runtime.h>

// ---------------- types ----------------
typedef __attribute__((ext_vector_type(16))) _Float16 v16h;
typedef __attribute__((ext_vector_type(8)))  _Float16 v8h;
typedef __attribute__((ext_vector_type(8)))  float    v8f;

#define BQ 8
#define SQ 128
#define EE 128
#define HH 512
#define AA 1024
#define VV 32000
#define OOV 64

__device__ __forceinline__ float sigf(float x) { return 1.0f / (1.0f + __expf(-x)); }

__device__ __forceinline__ v16h cat8(v8h lo, v8h hi) {
    v16h r;
#pragma unroll
    for (int i = 0; i < 8; ++i) { r[i] = lo[i]; r[i + 8] = hi[i]; }
    return r;
}

// ---------------- utility kernels ----------------
__global__ void zero_f4(float4* __restrict__ p, long n4) {
    for (long i = blockIdx.x * (long)blockDim.x + threadIdx.x; i < n4;
         i += (long)gridDim.x * blockDim.x) {
        float4 z; z.x = 0.f; z.y = 0.f; z.z = 0.f; z.w = 0.f;
        p[i] = z;
    }
}

__global__ void cvt_f32_f16(const float* __restrict__ s, _Float16* __restrict__ d, long n) {
    for (long i = blockIdx.x * (long)blockDim.x + threadIdx.x; i < n;
         i += (long)gridDim.x * blockDim.x)
        d[i] = (_Float16)s[i];
}

// Embedding gather: row = b*S+s.  Writes ex_f32 (b,s,E) and xs_h (s,b,E) for GEMM A.
__global__ void embed_gather(const int* __restrict__ x, const float* __restrict__ emb,
                             float* __restrict__ ex_f32, _Float16* __restrict__ xs_h) {
    int row = blockIdx.x;               // b*128+s
    int b = row >> 7, s = row & 127;
    int idx = x[row];
    float v = emb[(long)idx * EE + threadIdx.x];
    ex_f32[(long)row * EE + threadIdx.x] = v;
    xs_h[((long)(s * BQ + b)) * EE + threadIdx.x] = (_Float16)v;
}

// ---------------- generic WMMA GEMM ----------------
// C[m,n] = sum_k A[m,k] * W[n,k]  (+bias0[n]+bias1[n]).  A is f16 row-major (ld=K),
// W is row-major N x K, either f16 or f32 (converted to f16 while staging into LDS).
// Block tile 64x64, 8 waves, each wave: one 16-row tile x two 16-col tiles.
template <int W_F32>
__global__ __launch_bounds__(256) void gemm16(const _Float16* __restrict__ A,
                                              const void* __restrict__ Wv,
                                              const float* __restrict__ bias0,
                                              const float* __restrict__ bias1,
                                              float* __restrict__ C,
                                              int Mvalid, int K, int ldc, int rowmap,
                                              long sA, long sW, long sC) {
    const int bz = blockIdx.z;
    A += (long)bz * sA;
    C += (long)bz * sC;
    const _Float16* Wh = (const _Float16*)Wv + (W_F32 ? 0 : (long)bz * sW);
    const float*    Wf = (const float*)Wv    + (W_F32 ? (long)bz * sW : 0);

    __shared__ _Float16 As[64 * 32];
    __shared__ _Float16 Ws[64 * 32];

    const int m0 = blockIdx.x * 64;
    const int n0 = blockIdx.y * 64;
    const int tid  = threadIdx.x;
    const int wave = tid >> 5, lane = tid & 31;
    const int ll = lane & 15, h8 = lane >> 4;
    const int mt = wave & 3;        // m tile 0..3
    const int npair = wave >> 2;    // 0..1 -> n tiles npair*2, npair*2+1

    v8f acc0 = {}; v8f acc1 = {};

    const int lr = tid >> 2;        // 0..63 stage row
    const int lc = (tid & 3) * 8;   // 0,8,16,24 stage col

    for (int k0 = 0; k0 < K; k0 += 32) {
        // stage A
        {
            int gr = m0 + lr;
            v8h av = {};
            if (gr < Mvalid) av = *(const v8h*)(A + (long)gr * K + k0 + lc);
            *(v8h*)&As[lr * 32 + lc] = av;
        }
        // stage W (convert f32->f16 if needed)
        {
            int wr = n0 + lr;
            if (W_F32) {
                const float* src = Wf + (long)wr * K + k0 + lc;
                v8h wv;
#pragma unroll
                for (int i = 0; i < 8; ++i) wv[i] = (_Float16)src[i];
                *(v8h*)&Ws[lr * 32 + lc] = wv;
            } else {
                *(v8h*)&Ws[lr * 32 + lc] = *(const v8h*)(Wh + (long)wr * K + k0 + lc);
            }
        }
        __syncthreads();

        // A fragment (per-lane-half contiguous 16 K values)
        const _Float16* pa = &As[(mt * 16 + ll) * 32];
        v16h af = *(const v16h*)(pa + h8 * 16);
        // B fragments (per-lane-half contiguous 16 K values)
        const _Float16* pb0 = &Ws[(npair * 32 + ll) * 32];
        const _Float16* pb1 = &Ws[(npair * 32 + 16 + ll) * 32];
        v16h bf0 = *(const v16h*)(pb0 + h8 * 16);
        v16h bf1 = *(const v16h*)(pb1 + h8 * 16);

        acc0 = __builtin_amdgcn_wmma_f32_16x16x32_f16(false, af, false, bf0,
                                                      (short)0, acc0, false, false);
        acc1 = __builtin_amdgcn_wmma_f32_16x16x32_f16(false, af, false, bf1,
                                                      (short)0, acc1, false, false);
        __syncthreads();
    }

#pragma unroll
    for (int tile = 0; tile < 2; ++tile) {
        v8f acc = tile ? acc1 : acc0;
        int n = n0 + npair * 32 + tile * 16 + ll;
        float bv = (bias0 ? bias0[n] : 0.f) + (bias1 ? bias1[n] : 0.f);
#pragma unroll
        for (int r = 0; r < 8; ++r) {
            int m = m0 + mt * 16 + r + 8 * h8;
            if (m < Mvalid) {
                long mo = rowmap ? (long)(m + (m >> 7) + 1) : (long)m;
                C[mo * (long)ldc + n] = acc[r] + bv;
            }
        }
    }
}

// ---------------- LSTM recurrence step (3 LSTMs fused, 1 launch per t) ---------------
struct LstmOne {
    const float*    xproj;   // (S*B, 4H) includes bi+bh
    const _Float16* wh;      // (4H, H) f16
    const _Float16* hin;     // (16, H) f16  (rows 8..15 zero)
    _Float16*       hout;    // (16, H) f16  double buffer
    float*          c;       // (B, H)
    float*          hs;      // (S, B, H) f32 output
    int t_in;
    int t_store;
};
struct StepArgs { LstmOne l[3]; };

__global__ __launch_bounds__(256) void lstm_step(StepArgs args) {
    const LstmOne L = args.l[blockIdx.y];
    const int wg = blockIdx.x;                 // j range [wg*64, wg*64+64)
    __shared__ _Float16 Hs[16 * HH];           // 16 KB: padded h state
    __shared__ float G[4][64][8];              // gates staging (strip, jj, b)

    const int tid = threadIdx.x;
    for (int i = tid * 8; i < 16 * HH; i += 256 * 8)
        *(v8h*)&Hs[i] = *(const v8h*)&L.hin[i];
    __syncthreads();

    const int wave = tid >> 5, lane = tid & 31;
    const int ll = lane & 15, h8 = lane >> 4;
    const int tt0 = 2 * wave;                  // tiles tt0, tt0+1 of 16

    v8f acc[2] = {{}, {}};
    for (int k0 = 0; k0 < HH; k0 += 32) {
        const _Float16* pa = &Hs[ll * HH + k0];
        v16h af = *(const v16h*)(pa + h8 * 16);
#pragma unroll
        for (int u = 0; u < 2; ++u) {
            int tt = tt0 + u;
            int strip = tt >> 2, c16 = tt & 3;
            int gcol = strip * HH + wg * 64 + c16 * 16 + ll;   // row of wh
            const _Float16* pw = L.wh + (long)gcol * HH + k0;
            v16h bf = *(const v16h*)(pw + h8 * 16);
            acc[u] = __builtin_amdgcn_wmma_f32_16x16x32_f16(false, af, false, bf,
                                                            (short)0, acc[u], false, false);
        }
    }

#pragma unroll
    for (int u = 0; u < 2; ++u) {
        int tt = tt0 + u;
        int strip = tt >> 2, c16 = tt & 3;
        int nloc = c16 * 16 + ll;              // 0..63
        int gcol = strip * HH + wg * 64 + nloc;
#pragma unroll
        for (int r = 0; r < 8; ++r) {
            int m = r + 8 * h8;
            if (m < BQ)
                G[strip][nloc][m] = acc[u][r] +
                    L.xproj[((long)L.t_in * BQ + m) * (4 * HH) + gcol];
        }
    }
    __syncthreads();

    for (int it = tid; it < 64 * BQ; it += 256) {
        int jj = it >> 3, b = it & 7;
        int J = wg * 64 + jj;
        float ig = G[0][jj][b], fg = G[1][jj][b], gg = G[2][jj][b], og = G[3][jj][b];
        float co = L.c[b * HH + J];
        float cn = sigf(fg) * co + sigf(ig) * tanhf(gg);
        float hn = sigf(og) * tanhf(cn);
        L.c[b * HH + J] = cn;
        L.hs[((long)L.t_store * BQ + b) * HH + J] = hn;
        L.hout[b * HH + J] = (_Float16)hn;
    }
}

// ---------------- packing / small kernels ----------------
__global__ void pack_enc_outT(const float* __restrict__ hsF, const float* __restrict__ hsB,
                              _Float16* __restrict__ encT) {
    int bd = blockIdx.x;                 // b*1024 + d
    int b = bd >> 10, d = bd & 1023;
    int s = threadIdx.x;                 // 0..127
    float v = (d < HH) ? hsF[((long)s * BQ + b) * HH + d]
                       : hsB[((long)s * BQ + b) * HH + (d - HH)];
    encT[(long)bd * SQ + s] = (_Float16)v;
}

__global__ void pack_dec_h(const float* __restrict__ dec_hs, _Float16* __restrict__ dec_h_h) {
    int row = blockIdx.x;                // b*128+s
    int b = row >> 7, s = row & 127;
    for (int j = threadIdx.x; j < HH; j += 256)
        dec_h_h[(long)row * HH + j] = (_Float16)dec_hs[((long)s * BQ + b) * HH + j];
}

__global__ void pack_hcat(const _Float16* __restrict__ hF, const _Float16* __restrict__ hB,
                          _Float16* __restrict__ hcat) {
    int i = blockIdx.x * 256 + threadIdx.x;     // 16*1024
    int r = i >> 10, c = i & 1023;
    _Float16 v = (_Float16)0.f;
    if (r < BQ) v = (c < HH) ? hF[r * HH + c] : hB[r * HH + (c - HH)];
    hcat[i] = v;
}

__global__ void pad_state_h(const float* __restrict__ s, _Float16* __restrict__ d) {
    int i = blockIdx.x * 256 + threadIdx.x;     // 16*512
    int r = i >> 9;
    d[i] = (r < BQ) ? (_Float16)s[i] : (_Float16)0.f;
}

__global__ void pack_dec_in(const float* __restrict__ context, const _Float16* __restrict__ dec_h_h,
                            _Float16* __restrict__ dec_in) {
    int row = blockIdx.x;                // b*128+s
    for (int c = threadIdx.x; c < AA + HH; c += 256) {
        _Float16 v = (c < AA) ? (_Float16)context[(long)row * AA + c]
                              : dec_h_h[(long)row * HH + (c - AA)];
        dec_in[(long)row * (AA + HH) + c] = v;
    }
}

// attention: per (b,s) row: e = tanh(enc_feat[b]+dec_feat[b,s]); attn = v*softmax(e)
__global__ __launch_bounds__(256) void attn_kernel(const float* __restrict__ enc_feat,
                                                   const float* __restrict__ dec_feat,
                                                   const float* __restrict__ vvec,
                                                   float* __restrict__ attn0,
                                                   _Float16* __restrict__ attn_h) {
    int row = blockIdx.x;                // b*128+s
    int b = row >> 7;
    const float* df = dec_feat + (long)row * AA;
    const float* ef = enc_feat + (long)b * AA;
    __shared__ float red[256];
    __shared__ float ev[AA];
    int tid = threadIdx.x;

    float mx = -3.4e38f;
    for (int a = tid; a < AA; a += 256) {
        float e = tanhf(ef[a] + df[a]);
        ev[a] = e;
        mx = fmaxf(mx, e);
    }
    red[tid] = mx; __syncthreads();
    for (int st = 128; st > 0; st >>= 1) {
        if (tid < st) red[tid] = fmaxf(red[tid], red[tid + st]);
        __syncthreads();
    }
    mx = red[0]; __syncthreads();

    float sum = 0.f;
    for (int a = tid; a < AA; a += 256) {
        float e = __expf(ev[a] - mx);
        ev[a] = e;
        sum += e;
    }
    red[tid] = sum; __syncthreads();
    for (int st = 128; st > 0; st >>= 1) {
        if (tid < st) red[tid] += red[tid + st];
        __syncthreads();
    }
    float inv = 1.0f / red[0];

    for (int a = tid; a < AA; a += 256) {
        float p = vvec[a] * ev[a] * inv;
        if (a == 0) attn0[row] = p;
        if (a < SQ) attn_h[(long)row * SQ + a] = (_Float16)p;
    }
}

// p_gen + copy weight
__global__ __launch_bounds__(256) void pgen_kernel(const float* __restrict__ context,
                                                   const float* __restrict__ ex_f32,
                                                   const float* __restrict__ dec_hs,
                                                   const float* __restrict__ pg1,
                                                   const float* __restrict__ pg2,
                                                   const float* __restrict__ pg3,
                                                   const float* __restrict__ attn0,
                                                   float* __restrict__ pgen,
                                                   float* __restrict__ copyw) {
    int row = blockIdx.x;                // b*128+s
    int b = row >> 7, s = row & 127;
    __shared__ float red[256];
    int tid = threadIdx.x;
    float acc = 0.f;
    for (int a = tid; a < AA; a += 256) acc += context[(long)row * AA + a] * pg2[a];
    for (int a = tid; a < EE; a += 256) acc += ex_f32[(long)row * EE + a] * pg1[a];
    for (int a = tid; a < HH; a += 256) acc += dec_hs[((long)s * BQ + b) * HH + a] * pg3[a];
    red[tid] = acc; __syncthreads();
    for (int st = 128; st > 0; st >>= 1) {
        if (tid < st) red[tid] += red[tid + st];
        __syncthreads();
    }
    if (tid == 0) {
        float pg = sigf(red[0]);
        pgen[row] = pg;
        copyw[row] = (1.0f - pg) * attn0[row];
    }
}

// in-place softmax over V logits in d_out row, scale by p_gen, add copy weight at x
__global__ __launch_bounds__(256) void softmax_final(float* __restrict__ out,
                                                     const float* __restrict__ pgen,
                                                     const float* __restrict__ copyw,
                                                     const int* __restrict__ x) {
    int row = blockIdx.x;                // b*128+s
    int b = row >> 7, s = row & 127;
    float* r = out + ((long)(b * (SQ + 1) + s + 1)) * (VV + OOV);
    __shared__ float red[256];
    int tid = threadIdx.x;

    float mx = -3.4e38f;
    for (int i = tid; i < VV; i += 256) mx = fmaxf(mx, r[i]);
    red[tid] = mx; __syncthreads();
    for (int st = 128; st > 0; st >>= 1) {
        if (tid < st) red[tid] = fmaxf(red[tid], red[tid + st]);
        __syncthreads();
    }
    mx = red[0]; __syncthreads();

    float sum = 0.f;
    for (int i = tid; i < VV; i += 256) sum += __expf(r[i] - mx);
    red[tid] = sum; __syncthreads();
    for (int st = 128; st > 0; st >>= 1) {
        if (tid < st) red[tid] += red[tid + st];
        __syncthreads();
    }
    float scale = pgen[row] / red[0];

    for (int i = tid; i < VV; i += 256) r[i] = __expf(r[i] - mx) * scale;
    __syncthreads();
    if (tid == 0) r[x[row]] += copyw[row];
}

// ---------------- host ----------------
extern "C" void kernel_launch(void* const* d_in, const int* in_sizes, int n_in,
                              void* d_out, int out_size, void* d_ws, size_t ws_size,
                              hipStream_t stream) {
    (void)in_sizes; (void)n_in; (void)ws_size;

    const int*   x          = (const int*)d_in[0];
    const float* emb        = (const float*)d_in[4];
    const float* enc_wi_f   = (const float*)d_in[5];
    const float* enc_wh_f   = (const float*)d_in[6];
    const float* enc_bi_f   = (const float*)d_in[7];
    const float* enc_bh_f   = (const float*)d_in[8];
    const float* enc_wi_b   = (const float*)d_in[9];
    const float* enc_wh_b   = (const float*)d_in[10];
    const float* enc_bi_b   = (const float*)d_in[11];
    const float* enc_bh_b   = (const float*)d_in[12];
    const float* fc_state_w = (const float*)d_in[13];
    const float* fc_state_b = (const float*)d_in[14];
    const float* dec_wi     = (const float*)d_in[15];
    const float* dec_wh     = (const float*)d_in[16];
    const float* dec_bi     = (const float*)d_in[17];
    const float* dec_bh     = (const float*)d_in[18];
    const float* W_h_w      = (const float*)d_in[19];
    const float* W_h_b      = (const float*)d_in[20];
    const float* W_s_w      = (const float*)d_in[21];
    const float* W_s_b      = (const float*)d_in[22];
    const float* vvec       = (const float*)d_in[23];
    const float* fc1_w      = (const float*)d_in[24];
    const float* fc1_b      = (const float*)d_in[25];
    const float* fc2_w      = (const float*)d_in[26];
    const float* fc2_b      = (const float*)d_in[27];
    const float* pg1        = (const float*)d_in[28];
    const float* pg2        = (const float*)d_in[29];
    const float* pg3        = (const float*)d_in[30];
    float* out = (float*)d_out;

    // -------- workspace carve --------
    size_t off = 0;
    auto alloc = [&](size_t bytes) -> char* {
        off = (off + 255) & ~(size_t)255;
        char* r = (char*)d_ws + off;
        off += bytes;
        return r;
    };
    const int MR = BQ * SQ;           // 1024 rows
    _Float16* whF_h = (_Float16*)alloc((size_t)4 * HH * HH * 2);
    _Float16* whB_h = (_Float16*)alloc((size_t)4 * HH * HH * 2);
    _Float16* whD_h = (_Float16*)alloc((size_t)4 * HH * HH * 2);
    _Float16* wiF_h = (_Float16*)alloc((size_t)4 * HH * EE * 2);
    _Float16* wiB_h = (_Float16*)alloc((size_t)4 * HH * EE * 2);
    _Float16* wiD_h = (_Float16*)alloc((size_t)4 * HH * EE * 2);
    _Float16* fcsw_h = (_Float16*)alloc((size_t)HH * (2 * HH) * 2);
    _Float16* Whw_h = (_Float16*)alloc((size_t)AA * HH * 2);
    _Float16* Wsw_h = (_Float16*)alloc((size_t)AA * HH * 2);
    _Float16* fc1w_h = (_Float16*)alloc((size_t)(2 * HH) * (HH + AA) * 2);
    _Float16* xs_h = (_Float16*)alloc((size_t)MR * EE * 2);
    float* ex_f32 = (float*)alloc((size_t)MR * EE * 4);
    float* xprojF = (float*)alloc((size_t)MR * 4 * HH * 4);
    float* xprojB = (float*)alloc((size_t)MR * 4 * HH * 4);
    float* xprojD = (float*)alloc((size_t)MR * 4 * HH * 4);
    char* stateblk = alloc((size_t)6 * 16384 + (size_t)3 * 16384);
    _Float16* hbuf[3][2];
    float* cbuf[3];
    for (int l = 0; l < 3; ++l) {
        for (int par = 0; par < 2; ++par)
            hbuf[l][par] = (_Float16*)(stateblk + (size_t)(l * 2 + par) * 16384);
        cbuf[l] = (float*)(stateblk + (size_t)6 * 16384 + (size_t)l * 16384);
    }
    float* hsF = (float*)alloc((size_t)MR * HH * 4);
    float* hsB = (float*)alloc((size_t)MR * HH * 4);
    float* hsD = (float*)alloc((size_t)MR * HH * 4);
    _Float16* encT_h = (_Float16*)alloc((size_t)BQ * AA * SQ * 2);
    _Float16* dec_h_h = (_Float16*)alloc((size_t)MR * HH * 2);
    _Float16* hcat_h = (_Float16*)alloc((size_t)16 * (2 * HH) * 2);
    float* enc_state_f = (float*)alloc((size_t)16 * HH * 4);
    _Float16* enc_state_h = (_Float16*)alloc((size_t)16 * HH * 2);
    float* enc_feat = (float*)alloc((size_t)BQ * AA * 4);
    float* dec_feat = (float*)alloc((size_t)MR * AA * 4);
    float* attn0 = (float*)alloc((size_t)MR * 4);
    _Float16* attn_h = (_Float16*)alloc((size_t)MR * SQ * 2);
    float* context = (float*)alloc((size_t)MR * AA * 4);
    _Float16* dec_in_h = (_Float16*)alloc((size_t)MR * (AA + HH) * 2);
    float* fc1_f = (float*)alloc((size_t)MR * (2 * HH) * 4);
    _Float16* fc1_h = (_Float16*)alloc((size_t)MR * (2 * HH) * 2);
    float* pgen = (float*)alloc((size_t)MR * 4);
    float* copyw = (float*)alloc((size_t)MR * 4);

    // -------- zero output + recurrence state --------
    zero_f4<<<4096, 256, 0, stream>>>((float4*)out, (long)out_size / 4);
    zero_f4<<<36, 256, 0, stream>>>((float4*)stateblk, (long)(9 * 16384) / 16);

    // -------- weight conversions to f16 --------
    auto cvt = [&](const float* s, _Float16* d, long n) {
        long g = (n + 255) / 256; if (g > 4096) g = 4096;
        cvt_f32_f16<<<(int)g, 256, 0, stream>>>(s, d, n);
    };
    cvt(enc_wh_f, whF_h, (long)4 * HH * HH);
    cvt(enc_wh_b, whB_h, (long)4 * HH * HH);
    cvt(dec_wh,   whD_h, (long)4 * HH * HH);
    cvt(enc_wi_f, wiF_h, (long)4 * HH * EE);
    cvt(enc_wi_b, wiB_h, (long)4 * HH * EE);
    cvt(dec_wi,   wiD_h, (long)4 * HH * EE);
    cvt(fc_state_w, fcsw_h, (long)HH * 2 * HH);
    cvt(W_h_w, Whw_h, (long)AA * HH);
    cvt(W_s_w, Wsw_h, (long)AA * HH);
    cvt(fc1_w, fc1w_h, (long)(2 * HH) * (HH + AA));

    // -------- embedding gather --------
    embed_gather<<<MR, EE, 0, stream>>>(x, emb, ex_f32, xs_h);

    // -------- GEMM helper --------
    auto gemm = [&](const _Float16* A, const void* W, const float* b0, const float* b1,
                    float* C, int Mpad, int Mvalid, int N, int K, int ldc, int rowmap,
                    int batch, long sA, long sW, long sC, bool wf32) {
        dim3 g((Mpad + 63) / 64, N / 64, batch);
        if (wf32)
            gemm16<1><<<g, 256, 0, stream>>>(A, W, b0, b1, C, Mvalid, K, ldc, rowmap, sA, sW, sC);
        else
            gemm16<0><<<g, 256, 0, stream>>>(A, W, b0, b1, C, Mvalid, K, ldc, rowmap, sA, sW, sC);
    };

    // xproj = xs @ wi.T + (bi + bh), layout (s*B+b, 4H)
    gemm(xs_h, wiF_h, enc_bi_f, enc_bh_f, xprojF, MR, MR, 4 * HH, EE, 4 * HH, 0, 1, 0, 0, 0, false);
    gemm(xs_h, wiB_h, enc_bi_b, enc_bh_b, xprojB, MR, MR, 4 * HH, EE, 4 * HH, 0, 1, 0, 0, 0, false);
    gemm(xs_h, wiD_h, dec_bi,   dec_bh,   xprojD, MR, MR, 4 * HH, EE, 4 * HH, 0, 1, 0, 0, 0, false);

    // -------- LSTM recurrence: 128 stream-ordered step launches, 3 LSTMs fused --------
    for (int t = 0; t < SQ; ++t) {
        int pin = t & 1, pout = (t + 1) & 1;
        StepArgs sa;
        sa.l[0] = { xprojF, whF_h, hbuf[0][pin], hbuf[0][pout], cbuf[0], hsF, t, t };
        sa.l[1] = { xprojB, whB_h, hbuf[1][pin], hbuf[1][pout], cbuf[1], hsB, SQ - 1 - t, SQ - 1 - t };
        sa.l[2] = { xprojD, whD_h, hbuf[2][pin], hbuf[2][pout], cbuf[2], hsD, t, t };
        lstm_step<<<dim3(8, 3), 256, 0, stream>>>(sa);
    }
    // final states in parity 0 buffers (S even)
    _Float16* hF_fin = hbuf[0][0];
    _Float16* hB_fin = hbuf[1][0];

    // -------- packs --------
    pack_enc_outT<<<BQ * AA, SQ, 0, stream>>>(hsF, hsB, encT_h);
    pack_dec_h<<<MR, 256, 0, stream>>>(hsD, dec_h_h);
    pack_hcat<<<(16 * 2 * HH) / 256, 256, 0, stream>>>(hF_fin, hB_fin, hcat_h);

    // enc_state = hcat @ fc_state_w.T + b   (M=8 padded to 16)
    gemm(hcat_h, fcsw_h, fc_state_b, nullptr, enc_state_f, 16, BQ, HH, 2 * HH, HH, 0, 1, 0, 0, 0, false);
    pad_state_h<<<(16 * HH) / 256, 256, 0, stream>>>(enc_state_f, enc_state_h);

    // enc_feat = enc_state @ W_h_w.T + W_h_b
    gemm(enc_state_h, Whw_h, W_h_b, nullptr, enc_feat, 16, BQ, AA, HH, AA, 0, 1, 0, 0, 0, false);

    // dec_feat = dec_h @ W_s_w.T + W_s_b   (rows b*S+s)
    gemm(dec_h_h, Wsw_h, W_s_b, nullptr, dec_feat, MR, MR, AA, HH, AA, 0, 1, 0, 0, 0, false);

    // attention softmax
    attn_kernel<<<MR, 256, 0, stream>>>(enc_feat, dec_feat, vvec, attn0, attn_h);

    // context: per-batch  attn(128x128) @ encT(b).T   (encT rows = feature d, K = s)
    gemm(attn_h, encT_h, nullptr, nullptr, context, SQ, SQ, AA, SQ, AA, 0, BQ,
         (long)SQ * SQ, (long)AA * SQ, (long)SQ * AA, false);

    // dec_in = [context, dec_h]
    pack_dec_in<<<MR, 256, 0, stream>>>(context, dec_h_h, dec_in_h);

    // fc1
    gemm(dec_in_h, fc1w_h, fc1_b, nullptr, fc1_f, MR, MR, 2 * HH, AA + HH, 2 * HH, 0, 1, 0, 0, 0, false);
    cvt(fc1_f, fc1_h, (long)MR * 2 * HH);

    // logits -> d_out rows (b*129 + s + 1), fc2_w converted f32->f16 inline
    gemm(fc1_h, fc2_w, fc2_b, nullptr, out, MR, MR, VV, 2 * HH, VV + OOV, 1, 1, 0, 0, 0, true);

    // p_gen, copy weight, softmax + scatter (in place in d_out)
    pgen_kernel<<<MR, 256, 0, stream>>>(context, ex_f32, hsD, pg1, pg2, pg3, attn0, pgen, copyw);
    softmax_final<<<MR, 256, 0, stream>>>(out, pgen, copyw, x);
}
